// MultiQueryAttention_39616778338622
// MI455X (gfx1250) — matrix operand order repaired and checked by reading, not compiled, not run on hardware
//
#include <hip/hip_runtime.h>
#include <hip/hip_bf16.h>

// Problem constants (from reference): B=2, S=2048, D=1024, H=16, HD=64
#define CB 2
#define CS 2048
#define CD 1024
#define CH 16
#define CHD 64
#define CE (CD + 2 * CHD)   // 1152

typedef __attribute__((ext_vector_type(16))) __bf16 v16bf;
typedef __attribute__((ext_vector_type(8)))  float  v8f;

static __device__ __forceinline__ v8f wmma_bf16(v16bf a, v16bf b, v8f c) {
  // D(f32 16x16) = A(bf16 16x32) * B(bf16 32x16) + C
  return __builtin_amdgcn_wmma_f32_16x16x32_bf16(false, a, false, b,
                                                 (short)0, c, false, false);
}

// Put 4 consecutive fragment elements from a float4 into a v16bf.
#define SET4(vec, base, f4)              \
  do {                                   \
    (vec)[(base) + 0] = (__bf16)(f4).x;  \
    (vec)[(base) + 1] = (__bf16)(f4).y;  \
    (vec)[(base) + 2] = (__bf16)(f4).z;  \
    (vec)[(base) + 3] = (__bf16)(f4).w;  \
  } while (0)

// Build an A-fragment (16x32 bf16) from a row pointer at k-offset kb.
// Per-lane elements: k = kb + half*16 + j
static __device__ __forceinline__ v16bf frag_a(const float* rowp, int kb, int half) {
  const float4* p4 = (const float4*)(rowp + kb + (half << 4));
  float4 g0 = p4[0], g1 = p4[1];     // k = kb+half*16   .. +7
  float4 g2 = p4[2], g3 = p4[3];     // k = kb+half*16+8 .. +7
  v16bf v;
  SET4(v, 0, g0); SET4(v, 4, g1); SET4(v, 8, g2); SET4(v, 12, g3);
  return v;
}

// Build a B-fragment (32x16 bf16): per-lane elements k = kb + half*16 + j.
static __device__ __forceinline__ v16bf frag_b(const float* rowp, int kb, int half) {
  const float4* p4 = (const float4*)(rowp + kb + (half << 4));
  float4 g0 = p4[0], g1 = p4[1], g2 = p4[2], g3 = p4[3];
  v16bf v;
  SET4(v, 0, g0); SET4(v, 4, g1); SET4(v, 8, g2); SET4(v, 12, g3);
  return v;
}

// ---------------------------------------------------------------------------
// Generic C[M,N] = A[M,K] * W[N,K]^T + bias[N]   (f32 in memory, bf16 MACs)
// Block: 256 threads = 8 waves; block tile 16 rows x 128 cols.
// grid.x = M/16, grid.y = N/128.
// ---------------------------------------------------------------------------
__global__ __launch_bounds__(256)
void gemm_xwT_bias(const float* __restrict__ A, const float* __restrict__ W,
                   const float* __restrict__ bias, float* __restrict__ C,
                   int M, int N, int K) {
  const int t    = threadIdx.x;
  const int lane = t & 31;
  const int wv   = t >> 5;
  const int half = lane >> 4;
  const int lm   = lane & 15;

  const int m0 = blockIdx.x * 16;
  const int n0 = blockIdx.y * 128 + wv * 16;

  const float* arow = A + (size_t)(m0 + lm) * K;
  const float* wrow = W + (size_t)(n0 + lm) * K;

  v8f acc = {};
  for (int kb = 0; kb < K; kb += 32) {
    __builtin_prefetch((const void*)(wrow + kb + 128), 0, 0);
    v16bf av = frag_a(arow, kb, half);
    v16bf bv = frag_b(wrow, kb, half);
    acc = wmma_bf16(av, bv, acc);
  }
  const float bc = bias[n0 + lm];
#pragma unroll
  for (int r = 0; r < 8; ++r) {
    const int row = m0 + r + (half << 3);
    C[(size_t)row * N + (n0 + lm)] = acc[r] + bc;
  }
}

// ---------------------------------------------------------------------------
// Fused MQA core: one workgroup per (b, h, 16-query slab).
//   scores(16x2048) -> 128 KB LDS slab (CDNA5: 320 KB/WGP), softmax in LDS,
//   one coalesced float4 attn write, then ctx = P@V with per-wave K-split.
// ---------------------------------------------------------------------------
__global__ __launch_bounds__(256)
void mqa_attn(const float* __restrict__ qkv, float* __restrict__ attn,
              float* __restrict__ ctx) {
  extern __shared__ float sm[];
  float* P    = sm;                 // 16 * 2048 score/prob slab
  float* Qs   = P + 16 * CS;        // 16 * 64 scaled Q tile
  float* red  = Qs + 16 * CHD;      // 16 * 16 reduction scratch
  float* rowm = red + 256;          // 16 row maxima
  float* rsum = rowm + 16;          // 16 reciprocal row sums

  const int blk = blockIdx.x;               // ((b*H + h)*(S/16) + it)
  const int it  = blk & 127;
  const int h   = (blk >> 7) & 15;
  const int b   = blk >> 11;
  const int m0  = it * 16;

  const int t    = threadIdx.x;
  const int lane = t & 31;
  const int wv   = t >> 5;
  const int half = lane >> 4;
  const int lm   = lane & 15;

  // --- load Q tile (float4), pre-multiplied by 1/sqrt(HD) = 0.125 ---
  {
    const int m = t >> 4;                 // 16 rows, 16 float4 per row
    const int d = (t & 15) << 2;
    float4 q = *(const float4*)&qkv[(size_t)(b * CS + m0 + m) * CE + h * CHD + d];
    q.x *= 0.125f; q.y *= 0.125f; q.z *= 0.125f; q.w *= 0.125f;
    *(float4*)&Qs[m * CHD + d] = q;
  }
  __syncthreads();

  // --- A fragments (reused for every key tile of this wave) ---
  const float* qrow = Qs + lm * CHD;
  const v16bf a0 = frag_a(qrow, 0, half);
  const v16bf a1 = frag_a(qrow, 32, half);

  // --- scores: wave wv covers key columns [wv*256, wv*256+256) ---
  for (int tile = 0; tile < 16; ++tile) {
    const int colbase = wv * 256 + tile * 16;
    const float* krow = qkv + (size_t)(b * CS + colbase + lm) * CE + CD;
    __builtin_prefetch((const void*)(krow + 16 * CE), 0, 0);  // next tile's K row
    v16bf b0 = frag_b(krow, 0, half);
    v16bf b1 = frag_b(krow, 32, half);
    v8f c = {};
    c = wmma_bf16(a0, b0, c);
    c = wmma_bf16(a1, b1, c);
#pragma unroll
    for (int r = 0; r < 8; ++r)
      P[(r + (half << 3)) * CS + colbase + lm] = c[r];
  }
  __syncthreads();

  // --- softmax over each row of the 16x2048 slab ---
  const int rr = t >> 4, cc = t & 15;
  float mx = -3.4e38f;
  for (int col = cc; col < CS; col += 16) mx = fmaxf(mx, P[rr * CS + col]);
  red[rr * 16 + cc] = mx;
  __syncthreads();
  if (t < 16) {
    float m2 = red[t * 16];
#pragma unroll
    for (int i = 1; i < 16; ++i) m2 = fmaxf(m2, red[t * 16 + i]);
    rowm[t] = m2;
  }
  __syncthreads();
  const float rm = rowm[rr];
  float s = 0.f;
  for (int col = cc; col < CS; col += 16) {
    const float e = __expf(P[rr * CS + col] - rm);
    P[rr * CS + col] = e;
    s += e;
  }
  red[rr * 16 + cc] = s;
  __syncthreads();
  if (t < 16) {
    float s2 = 0.f;
#pragma unroll
    for (int i = 0; i < 16; ++i) s2 += red[t * 16 + i];
    rsum[t] = 1.0f / s2;
  }
  __syncthreads();

  // --- normalize in LDS + coalesced float4 attn write (the HBM bottleneck) ---
  float* ap = attn + ((size_t)(b * CH + h) * CS + m0) * CS;
  for (int r = 0; r < 16; ++r) {
    const float ri = rsum[r];
#pragma unroll
    for (int col = (t << 2); col < CS; col += 1024) {
      float4 p = *(float4*)&P[r * CS + col];
      p.x *= ri; p.y *= ri; p.z *= ri; p.w *= ri;
      *(float4*)&P[r * CS + col] = p;
      *(float4*)&ap[(size_t)r * CS + col] = p;
    }
  }
  __syncthreads();

  // --- ctx(16x64) = P(16x2048) @ V(2048x64): wave wv takes K in [wv*256,+256) ---
  v8f acc[4];
  const v8f zero = {};
#pragma unroll
  for (int nt = 0; nt < 4; ++nt) acc[nt] = zero;

  const int kb0 = wv * 256;
  const float* prow = P + lm * CS;
  for (int ks = 0; ks < 8; ++ks) {
    const int kb = kb0 + ks * 32;
    v16bf pa = frag_a(prow, kb, half);     // ds_load_b128 feeds
#pragma unroll
    for (int nt = 0; nt < 4; ++nt) {
      v16bf bv;
#pragma unroll
      for (int j = 0; j < 16; ++j) {
        const int k = kb + (half << 4) + j;
        bv[j] = (__bf16)qkv[(size_t)(b * CS + k) * CE + (CD + CHD) + nt * 16 + lm];
      }
      acc[nt] = wmma_bf16(pa, bv, acc[nt]);
    }
  }
  __syncthreads();   // all waves done reading P before reuse

  // partial ctx per wave into reused P region: [wv][m][d]
#pragma unroll
  for (int nt = 0; nt < 4; ++nt)
#pragma unroll
    for (int r = 0; r < 8; ++r)
      P[wv * 1024 + (r + (half << 3)) * CHD + nt * 16 + lm] = acc[nt][r];
  __syncthreads();

  // reduce 8 wave partials (float4) and write ctx
  {
    const int m = t >> 4;
    const int d = (t & 15) << 2;
    float4 ssum = *(float4*)&P[m * CHD + d];
#pragma unroll
    for (int w = 1; w < 8; ++w) {
      float4 pp = *(float4*)&P[w * 1024 + m * CHD + d];
      ssum.x += pp.x; ssum.y += pp.y; ssum.z += pp.z; ssum.w += pp.w;
    }
    *(float4*)&ctx[(size_t)(b * CS + m0 + m) * CD + h * CHD + d] = ssum;
  }
}

// ---------------------------------------------------------------------------
extern "C" void kernel_launch(void* const* d_in, const int* in_sizes, int n_in,
                              void* d_out, int out_size, void* d_ws, size_t ws_size,
                              hipStream_t stream) {
  (void)in_sizes; (void)n_in; (void)out_size; (void)ws_size;

  const float* x      = (const float*)d_in[0];   // (B,S,D)
  const float* Wqkv_w = (const float*)d_in[1];   // (E,D)
  const float* Wqkv_b = (const float*)d_in[2];   // (E)
  const float* out_w  = (const float*)d_in[3];   // (D,D)
  const float* out_b  = (const float*)d_in[4];   // (D)

  float* out  = (float*)d_out;                              // (B,S,D)
  float* attn = out + (size_t)CB * CS * CD;                 // (B,H,S,S)

  float* qkv = (float*)d_ws;                                // (B,S,E)
  float* ctx = qkv + (size_t)CB * CS * CE;                  // (B,S,D)

  const int BS = CB * CS;  // 4096

  // 1) QKV projection: qkv = x @ Wqkv_w^T + Wqkv_b
  gemm_xwT_bias<<<dim3(BS / 16, CE / 128), 256, 0, stream>>>(
      x, Wqkv_w, Wqkv_b, qkv, BS, CE, CD);

  // 2) fused scores/softmax/attn-write/PV
  const size_t smemB = (size_t)(16 * CS + 16 * CHD + 256 + 32) * sizeof(float);
  hipFuncSetAttribute(reinterpret_cast<const void*>(mqa_attn),
                      hipFuncAttributeMaxDynamicSharedMemorySize, (int)smemB);
  mqa_attn<<<dim3(CB * CH * (CS / 16)), 256, smemB, stream>>>(qkv, attn, ctx);

  // 3) output projection: out = ctx @ out_w^T + out_b
  gemm_xwT_bias<<<dim3(BS / 16, CD / 128), 256, 0, stream>>>(
      ctx, out_w, out_b, out, BS, CD, CD);
}
